// CausalMultiHeadSelfAttention_65420941853259
// MI455X (gfx1250) — matrix operand order repaired and checked by reading, not compiled, not run on hardware
//
#include <hip/hip_runtime.h>
#include <math.h>
#include <stdint.h>

typedef unsigned short u16;
typedef unsigned int   u32;

typedef __attribute__((ext_vector_type(16))) __bf16 v16bf;
typedef __attribute__((ext_vector_type(8)))  float  v8f;
typedef __attribute__((ext_vector_type(4)))  unsigned int v4u;
typedef __attribute__((ext_vector_type(8)))  int v8i;
typedef __attribute__((ext_vector_type(4)))  int v4i;

#define B_    2
#define SEQ_  2048
#define DM_   2048
#define H_    16
#define DK_   128
#define SCALE_ 0.08838834764831845f   // 1/sqrt(128)
#define LOG_THETA 9.210340371976184f  // ln(10000)

// ---------- bf16 helpers (RNE) ----------
static __device__ inline u16 f2bf(float f) {
  u32 u = __float_as_uint(f);
  u32 r = (u + 0x7FFFu + ((u >> 16) & 1u)) >> 16;
  return (u16)r;
}
static __device__ inline float bf2f(u16 h) {
  return __uint_as_float(((u32)h) << 16);
}

union FragU { v16bf v; uint4 q[2]; };
union AccU  { v8f v; float f[8]; };

static __device__ inline v8f zero8() {
  v8f z = {0.f, 0.f, 0.f, 0.f, 0.f, 0.f, 0.f, 0.f};
  return z;
}

// A fragment 16x32 bf16 (row = lane%16, K split per CDNA5 layout), row stride in u16
static __device__ inline v16bf load_a_frag(const u16* base, int stride, int lane) {
  int r = lane & 15, g = lane >> 4;
  FragU f;
  f.q[0] = *(const uint4*)(base + (size_t)r * stride + 8 * g);        // K = 8g .. 8g+7
  f.q[1] = *(const uint4*)(base + (size_t)r * stride + 16 + 8 * g);   // K = 16+8g .. 16+8g+7
  return f.v;
}
// B fragment 32x16 bf16 (N = lane%16, K pairs packed per dword), memory is [N][K]
static __device__ inline v16bf load_b_frag(const u16* base, int stride, int lane) {
  int n = lane & 15, g = lane >> 4;
  FragU f;
  f.q[0] = *(const uint4*)(base + (size_t)n * stride + 8 * g);        // K = 8g .. 8g+7
  f.q[1] = *(const uint4*)(base + (size_t)n * stride + 16 + 8 * g);   // K = 16+8g .. 16+8g+7
  return f.v;
}

static __device__ inline v8f wmma_bf16(v16bf a, v16bf b, v8f c) {
  return __builtin_amdgcn_wmma_f32_16x16x32_bf16(false, a, false, b, (short)0, c,
                                                 false, false);
}

// ---------- TDM: 2D tile load (rows x DK_ bf16), LDS rows padded +16B ----------
// D# per CDNA5 ISA:
//  group0: [1:0]=count=1, [63:32]=lds_addr, [120:64]=global_addr, [127:126]=type=2
//  group1: data_size=1(2B), pad_enable=1, pad_interval=5 (64 dwords = 256B = one
//          128-elem row), pad_amount=3 (4 dwords = 16B -> DKP=136 u16 stride),
//          tensor_dim0=128, tensor_dim1=rows_remaining, tile 128x32, stride0=128
static __device__ inline void tdm_load_ktile(u32 lds_off, const void* gptr,
                                             int rows_remaining) {
  unsigned long long ga = (unsigned long long)(uintptr_t)gptr;
  v4u g0;
  g0.x = 1u;                             // count = 1 (valid user descriptor)
  g0.y = lds_off;                        // LDS byte address
  g0.z = (u32)ga;                        // global addr [31:0]
  g0.w = (u32)(ga >> 32) | (2u << 30);   // global addr [56:32] | type=2
  v8i g1;
  g1[0] = (1 << 16) | (1 << 20) | (5 << 22) | (3 << 25);  // dsize|pad_en|int|amt
  g1[1] = (DK_ & 0xffff) << 16;                           // tensor_dim0 lo16
  g1[2] = (rows_remaining & 0xffff) << 16;                // dim0 hi | tensor_dim1 lo
  g1[3] = (DK_ << 16);                                    // dim1 hi | tile_dim0=128
  g1[4] = 32;                                             // tile_dim1=32, tile_dim2=0
  g1[5] = DK_;                                            // tensor_dim0_stride lo32
  g1[6] = 0;
  g1[7] = 0;
  v4i z4 = {0, 0, 0, 0};
#if __clang_major__ >= 23
  v8i z8 = {0, 0, 0, 0, 0, 0, 0, 0};
  __builtin_amdgcn_tensor_load_to_lds(g0, g1, z4, z4, z8, 0);
#else
  __builtin_amdgcn_tensor_load_to_lds(g0, g1, z4, z4, 0);
#endif
}

// =====================================================================
// GEMM: out[m,n] = sum_k A[m,k] * W[n,k]   (A: [4096][2048], W: [2048][2048])
// split-bf16: A*W ~= Ahi*Whi + Ahi*Wlo + Alo*Whi  (3 bf16 WMMAs per tile)
// MODE 0: write fp32 to out_f
// MODE 1: RoPE, then bf16 hi/lo planes at [b,h,s,dk]
// MODE 2: bf16 hi plane only (V)
// =====================================================================
#define TM 128
#define TN 128
#define TK 32
#define KP 40   // u16 row stride (80B, 16B-aligned)

template <int MODE>
__global__ void __launch_bounds__(256)
proj_gemm(const float* __restrict__ A, const float* __restrict__ W,
          const int* __restrict__ pos, u16* __restrict__ out_hi,
          u16* __restrict__ out_lo, float* __restrict__ out_f) {
  __shared__ __align__(16) u16 a_hi[TM * KP];
  __shared__ __align__(16) u16 a_lo[TM * KP];
  __shared__ __align__(16) u16 w_hi[TN * KP];
  __shared__ __align__(16) u16 w_lo[TN * KP];

  const int tid = threadIdx.x, lane = tid & 31, wave = tid >> 5;
  const int m0 = blockIdx.x * TM, n0 = blockIdx.y * TN;

  v8f acc[8];
#pragma unroll
  for (int t = 0; t < 8; ++t) acc[t] = zero8();

  for (int k0 = 0; k0 < DM_; k0 += TK) {
#pragma unroll
    for (int i = 0; i < 4; ++i) {
      int idx = tid + 256 * i;          // 0..1023
      int row = idx >> 3, c4 = (idx & 7) * 4;
      const float* ap = A + (size_t)(m0 + row) * DM_ + k0 + c4;
      const float* wp = W + (size_t)(n0 + row) * DM_ + k0 + c4;
      // pull next K-slab toward the WGP while this tile computes
      if ((idx & 7) == 0 && k0 + TK < DM_) {
        __builtin_prefetch(ap + TK, 0, 1);
        __builtin_prefetch(wp + TK, 0, 1);
      }
      float4 av = *(const float4*)ap;
      ushort4 hi, lo;
      hi.x = f2bf(av.x); lo.x = f2bf(av.x - bf2f(hi.x));
      hi.y = f2bf(av.y); lo.y = f2bf(av.y - bf2f(hi.y));
      hi.z = f2bf(av.z); lo.z = f2bf(av.z - bf2f(hi.z));
      hi.w = f2bf(av.w); lo.w = f2bf(av.w - bf2f(hi.w));
      *(ushort4*)(a_hi + row * KP + c4) = hi;
      *(ushort4*)(a_lo + row * KP + c4) = lo;
      float4 wv = *(const float4*)wp;
      hi.x = f2bf(wv.x); lo.x = f2bf(wv.x - bf2f(hi.x));
      hi.y = f2bf(wv.y); lo.y = f2bf(wv.y - bf2f(hi.y));
      hi.z = f2bf(wv.z); lo.z = f2bf(wv.z - bf2f(hi.z));
      hi.w = f2bf(wv.w); lo.w = f2bf(wv.w - bf2f(hi.w));
      *(ushort4*)(w_hi + row * KP + c4) = hi;
      *(ushort4*)(w_lo + row * KP + c4) = lo;
    }
    __syncthreads();

    v16bf Ah = load_a_frag(a_hi + 16 * wave * KP, KP, lane);
    v16bf Al = load_a_frag(a_lo + 16 * wave * KP, KP, lane);
#pragma unroll
    for (int t = 0; t < 8; ++t) {
      v16bf Bh = load_b_frag(w_hi + 16 * t * KP, KP, lane);
      v16bf Bl = load_b_frag(w_lo + 16 * t * KP, KP, lane);
      acc[t] = wmma_bf16(Ah, Bh, acc[t]);
      acc[t] = wmma_bf16(Ah, Bl, acc[t]);
      acc[t] = wmma_bf16(Al, Bh, acc[t]);
    }
    __syncthreads();
  }

  // epilogue: C layout -> row = r + 8*(lane/16), col = 16t + lane%16
  const int g = lane >> 4, ln = lane & 15;
#pragma unroll
  for (int t = 0; t < 8; ++t) {
    int col = n0 + 16 * t + ln;
#pragma unroll
    for (int r = 0; r < 8; ++r) {
      int mrow = m0 + 16 * wave + r + 8 * g;
      float v = acc[t][r];
      if (MODE == 0) {
        out_f[(size_t)mrow * DM_ + col] = v;
      } else {
        int b = mrow / SEQ_, s = mrow % SEQ_;
        int hh = col / DK_, dk = col % DK_;
        float res = v;
        if (MODE == 1) {
          // rotary pair lives in the adjacent lane (same C-layout row)
          float partner = __shfl_xor(v, 1, 32);
          float p = (float)pos[s];
          float inv = __expf(-LOG_THETA * (float)(dk & ~1) * (1.0f / (float)DK_));
          float ang = p * inv;
          float c, sn;
          __sincosf(ang, &sn, &c);
          res = (dk & 1) ? (partner * sn + v * c) : (v * c - partner * sn);
        }
        size_t o = (((size_t)b * H_ + hh) * SEQ_ + s) * DK_ + dk;
        u16 hb = f2bf(res);
        out_hi[o] = hb;
        if (MODE == 1) out_lo[o] = f2bf(res - bf2f(hb));
      }
    }
  }
}

// =====================================================================
// Flash attention: grid (S/64, H, B), 128 threads = 4 waves,
// each wave owns 16 query rows x DK=128. QK^T in split-bf16, PV in bf16.
// K hi/lo tiles staged by the Tensor Data Mover; V staged transposed by VALU.
// =====================================================================
#define DKP 136  // K-tile row stride in u16 (272B) == TDM pad 256B+16B
#define VP  40   // V^T row stride in u16 (80B)

__global__ void __launch_bounds__(128)
attn_fwd(const u16* __restrict__ qhi, const u16* __restrict__ qlo,
         const u16* __restrict__ khi, const u16* __restrict__ klo,
         const u16* __restrict__ vpl, float* __restrict__ attn_out) {
  __shared__ __align__(16) u16 ks_hi[32 * DKP];
  __shared__ __align__(16) u16 ks_lo[32 * DKP];
  __shared__ __align__(16) u16 vt[DK_ * VP];        // V^T: [dk][key]
  __shared__ __align__(16) u16 p_buf[4][16 * 32];   // per-wave P bounce

  const int tid = threadIdx.x, lane = tid & 31, wave = tid >> 5;
  const int h = blockIdx.y, b = blockIdx.z;
  const int q0b = blockIdx.x * 64;
  const int q0w = q0b + 16 * wave;
  const size_t head_off = ((size_t)b * H_ + h) * SEQ_ * DK_;
  const int g = lane >> 4, ln = lane & 15;

  // resident Q fragments (hi/lo), 4 chunks of K=32 over DK=128
  v16bf Qh[4], Ql[4];
#pragma unroll
  for (int c = 0; c < 4; ++c) {
    Qh[c] = load_a_frag(qhi + head_off + (size_t)q0w * DK_ + 32 * c, DK_, lane);
    Ql[c] = load_a_frag(qlo + head_off + (size_t)q0w * DK_ + 32 * c, DK_, lane);
  }

  v8f oacc[8];
#pragma unroll
  for (int n = 0; n < 8; ++n) oacc[n] = zero8();
  float mst[8], lst[8];
#pragma unroll
  for (int r = 0; r < 8; ++r) { mst[r] = -3.0e38f; lst[r] = 0.f; }

  const int jend = q0b + 64;  // block-uniform causal bound
  for (int j0 = 0; j0 < jend; j0 += 32) {
    // --- K hi/lo via Tensor Data Mover (one DMA per plane, wave 0 only) ---
    if (wave == 0) {
      tdm_load_ktile((u32)(uintptr_t)ks_hi,
                     khi + head_off + (size_t)j0 * DK_, SEQ_ - j0);
      tdm_load_ktile((u32)(uintptr_t)ks_lo,
                     klo + head_off + (size_t)j0 * DK_, SEQ_ - j0);
    }
    // --- V staged transposed by all waves ---
#pragma unroll
    for (int i = 0; i < 4; ++i) {
      int idx = tid + 128 * i;          // 0..511
      int row = idx >> 4, c8 = (idx & 15) * 8;
      uint4 vv = *(const uint4*)(vpl + head_off + (size_t)(j0 + row) * DK_ + c8);
      const u16* ve = (const u16*)&vv;
#pragma unroll
      for (int e = 0; e < 8; ++e) vt[(size_t)(c8 + e) * VP + row] = ve[e];
    }
    if (wave == 0) __builtin_amdgcn_s_wait_tensorcnt(0);
    __syncthreads();

    if (j0 <= q0w + 15) {   // this wave has unmasked keys in this block
      AccU s0, s1;
      s0.v = zero8();
      s1.v = zero8();
#pragma unroll
      for (int c = 0; c < 4; ++c) {
        v16bf B0h = load_b_frag(ks_hi + 32 * c, DKP, lane);
        v16bf B0l = load_b_frag(ks_lo + 32 * c, DKP, lane);
        v16bf B1h = load_b_frag(ks_hi + 16 * DKP + 32 * c, DKP, lane);
        v16bf B1l = load_b_frag(ks_lo + 16 * DKP + 32 * c, DKP, lane);
        s0.v = wmma_bf16(Qh[c], B0h, s0.v);
        s0.v = wmma_bf16(Qh[c], B0l, s0.v);
        s0.v = wmma_bf16(Ql[c], B0h, s0.v);
        s1.v = wmma_bf16(Qh[c], B1h, s1.v);
        s1.v = wmma_bf16(Qh[c], B1l, s1.v);
        s1.v = wmma_bf16(Ql[c], B1h, s1.v);
      }
      // online softmax per C-layout row (16-lane groups)
      float p0[8], p1[8];
#pragma unroll
      for (int r = 0; r < 8; ++r) {
        int qrow = q0w + r + 8 * g;
        float x0 = s0.f[r] * SCALE_;
        float x1 = s1.f[r] * SCALE_;
        if (j0 + ln > qrow) x0 = -3.0e38f;
        if (j0 + 16 + ln > qrow) x1 = -3.0e38f;
        float rm = fmaxf(x0, x1);
#pragma unroll
        for (int m = 1; m < 16; m <<= 1) rm = fmaxf(rm, __shfl_xor(rm, m, 32));
        float mn = fmaxf(mst[r], rm);
        float alpha = __expf(mst[r] - mn);
        float e0 = __expf(x0 - mn);
        float e1 = __expf(x1 - mn);
        float rs = e0 + e1;
#pragma unroll
        for (int m = 1; m < 16; m <<= 1) rs += __shfl_xor(rs, m, 32);
        lst[r] = lst[r] * alpha + rs;
        mst[r] = mn;
#pragma unroll
        for (int n = 0; n < 8; ++n) oacc[n][r] *= alpha;
        p0[r] = e0;
        p1[r] = e1;
      }
      // bounce P through wave-private LDS to re-lay C -> A fragment
      u16* pb = &p_buf[wave][0];
#pragma unroll
      for (int r = 0; r < 8; ++r) {
        int row = r + 8 * g;
        pb[row * 32 + ln] = f2bf(p0[r]);
        pb[row * 32 + 16 + ln] = f2bf(p1[r]);
      }
      v16bf Pf = load_a_frag(pb, 32, lane);
#pragma unroll
      for (int n = 0; n < 8; ++n) {
        v16bf Bv = load_b_frag(vt + 16 * n * VP, VP, lane);
        oacc[n] = wmma_bf16(Pf, Bv, oacc[n]);
      }
    }
    __syncthreads();
  }

  // normalize + write fp32 [b][s][h*DK+dk]
#pragma unroll
  for (int n = 0; n < 8; ++n) {
    int dk = 16 * n + ln;
#pragma unroll
    for (int r = 0; r < 8; ++r) {
      int s = q0w + r + 8 * g;
      attn_out[((size_t)b * SEQ_ + s) * DM_ + h * DK_ + dk] = oacc[n][r] / lst[r];
    }
  }
}

// =====================================================================
extern "C" void kernel_launch(void* const* d_in, const int* in_sizes, int n_in,
                              void* d_out, int out_size, void* d_ws, size_t ws_size,
                              hipStream_t stream) {
  const float* x  = (const float*)d_in[0];
  const int*  pos = (const int*)d_in[1];
  const float* Wq = (const float*)d_in[2];
  const float* Wk = (const float*)d_in[3];
  const float* Wv = (const float*)d_in[4];
  const float* Wo = (const float*)d_in[5];

  const size_t NP = (size_t)B_ * H_ * SEQ_ * DK_;  // 8,388,608 elems per plane
  u16* qhi = (u16*)d_ws;
  u16* qlo = qhi + NP;
  u16* khi = qlo + NP;
  u16* klo = khi + NP;
  u16* vpl = klo + NP;
  float* attn = (float*)(vpl + NP);  // fp32 [B,S,D]

  dim3 gg((B_ * SEQ_) / TM, DM_ / TN, 1);  // (32, 16)
  proj_gemm<1><<<gg, 256, 0, stream>>>(x, Wq, pos, qhi, qlo, nullptr);
  proj_gemm<1><<<gg, 256, 0, stream>>>(x, Wk, pos, khi, klo, nullptr);
  proj_gemm<2><<<gg, 256, 0, stream>>>(x, Wv, pos, vpl, nullptr, nullptr);

  attn_fwd<<<dim3(SEQ_ / 64, H_, B_), 128, 0, stream>>>(qhi, qlo, khi, klo, vpl, attn);

  proj_gemm<0><<<gg, 256, 0, stream>>>(attn, Wo, nullptr, nullptr, nullptr,
                                       (float*)d_out);
}
